// Attention_68461778698771
// MI455X (gfx1250) — compile-verified, matrix operand order checked by reading, not run on hardware
//
#include <hip/hip_runtime.h>

// ---------------------------------------------------------------------------
// Multi-query causal attention for MI455X (gfx1250), all GEMM stages on
// V_WMMA_F32_16X16X32_BF16 (wave32). fp32 inputs converted to bf16 (fp32
// accumulation); weights pre-transposed N-major so A and B fragments are two
// contiguous b128 loads per lane. Flash attention: causal KV loop split into
// an unmasked steady-state phase and a masked diagonal/tail phase; row sums
// via WMMA against a ones matrix; P converted C-layout->A-layout through a
// per-wave LDS region fenced with s_wait_dscnt (no block barrier).
// ---------------------------------------------------------------------------

typedef __bf16 bf16;
typedef __attribute__((ext_vector_type(16))) __bf16 v16bf;
typedef __attribute__((ext_vector_type(8)))  __bf16 v8bf;
typedef __attribute__((ext_vector_type(8)))  float  v8f;

#define WMMA_BF16(a, b, c) \
  __builtin_amdgcn_wmma_f32_16x16x32_bf16(false, (a), false, (b), (short)0, (c), false, false)

#define NPAD 2080   // kv columns padded: 1 null + 2048 tokens + zero pad, multiple of 32

// 16x32 bf16 A/B fragment from a K-contiguous (row-major in K) matrix.
// ISA layout: lanes 0-15 hold row `r`, K = {k0..k0+7, k0+16..k0+23};
// lanes 16-31 hold row `r`, K shifted by +8. Two 16B loads per lane.
__device__ __forceinline__ v16bf load_frag(const bf16* p, int ld, int row, int k0, int half) {
  const bf16* base = p + (size_t)row * ld + k0 + half * 8;
  v8bf lo = *(const v8bf*)(base);
  v8bf hi = *(const v8bf*)(base + 16);
  v16bf f;
#pragma unroll
  for (int i = 0; i < 8; ++i) { f[i] = lo[i]; f[8 + i] = hi[i]; }
  return f;
}

// --------------------------- conversion kernels ----------------------------

__global__ void cvt_f32_bf16(const float* __restrict__ s, bf16* __restrict__ d, long n) {
  long i = (long)blockIdx.x * blockDim.x + threadIdx.x;
  if (i < n) d[i] = (bf16)s[i];
}

// Wt[n*K + k] = W[k*N + n] * scale   (N-major "Bt" layout for WMMA B operand)
__global__ void transpose_cvt(const float* __restrict__ W, bf16* __restrict__ Wt,
                              int K, int N, float scale) {
  int i = blockIdx.x * blockDim.x + threadIdx.x;
  if (i >= K * N) return;
  int k = i / N, n = i % N;
  Wt[(size_t)n * K + k] = (bf16)(W[i] * scale);
}

// Build kv [B][NPAD][64] (null row at j=0, zero pad j>Ntok) and kvT [B][64][NPAD]
__global__ void repack_kv(const bf16* __restrict__ kvtmp, const float* __restrict__ nullkv,
                          bf16* __restrict__ kvb, bf16* __restrict__ kvT, int B, int Ntok) {
  int i = blockIdx.x * blockDim.x + threadIdx.x;
  int total = B * NPAD * 64;
  if (i >= total) return;
  int d = i & 63;
  int j = (i >> 6) % NPAD;
  int b = i / (NPAD * 64);
  bf16 v;
  if (j == 0)            v = (bf16)nullkv[d];
  else if (j <= Ntok)    v = kvtmp[((size_t)b * Ntok + (j - 1)) * 64 + d];
  else                   v = (bf16)0.0f;
  kvb[((size_t)b * NPAD + j) * 64 + d]  = v;
  kvT[((size_t)b * 64 + d) * NPAD + j]  = v;
}

// ------------------------------ WMMA GEMM ----------------------------------
// C[M,N] = A[M,K] @ Bt[N,K]^T.  4 waves / block; wave w -> rows m0..m0+31,
// cols n0..n0+63: 2 A-fragments x 4 B-fragments -> 8 WMMAs per K-step of 32.
template <typename OutT>
__global__ __launch_bounds__(128) void gemm_bf16(const bf16* __restrict__ A,
                                                 const bf16* __restrict__ Bt,
                                                 OutT* __restrict__ C,
                                                 int M, int N, int K) {
  const int lane = threadIdx.x & 31;
  const int wave = threadIdx.x >> 5;
  const int r = lane & 15, half = lane >> 4;
  const int m0 = blockIdx.y * 128 + wave * 32;
  const int n0 = blockIdx.x * 64;

  v8f acc[2][4] = {};
  for (int k0 = 0; k0 < K; k0 += 32) {
    v16bf a0 = load_frag(A, K, m0 + r,      k0, half);
    v16bf a1 = load_frag(A, K, m0 + 16 + r, k0, half);
    if (k0 + 32 < K) {  // prefetch next K-tile (global_prefetch_b8)
      __builtin_prefetch(A  + (size_t)(m0 + r) * K + k0 + 32, 0, 3);
      __builtin_prefetch(Bt + (size_t)(n0 + r) * K + k0 + 32, 0, 3);
    }
#pragma unroll
    for (int t = 0; t < 4; ++t) {
      v16bf b = load_frag(Bt, K, n0 + t * 16 + r, k0, half);
      acc[0][t] = WMMA_BF16(a0, b, acc[0][t]);
      acc[1][t] = WMMA_BF16(a1, b, acc[1][t]);
    }
  }
#pragma unroll
  for (int mi = 0; mi < 2; ++mi)
#pragma unroll
    for (int t = 0; t < 4; ++t)
#pragma unroll
      for (int v = 0; v < 8; ++v) {
        int row = m0 + mi * 16 + v + 8 * half;  // C layout: VGPR v holds M = v + 8*half
        int col = n0 + t * 16 + r;              // lane r holds N = r
        C[(size_t)row * N + col] = (OutT)acc[mi][t][v];
      }
}

// --------------------------- flash attention -------------------------------
// One 32-kv-column step. MASKED=false is the steady-state (strictly below the
// causal diagonal, no bounds checks); MASKED=true handles diagonal + tail.
template <bool MASKED>
__device__ __forceinline__ void attn_step(int j0, int q0, int r, int half, int wave, int Ntok,
                                          const bf16* kvb, const bf16* kvt,
                                          const v16bf& qa0, const v16bf& qa1, const v16bf& ones,
                                          v8f o[4], float mrow[8], float lrow[8],
                                          bf16 (*plds)[16][32]) {
  v8f s[2];
#pragma unroll
  for (int t = 0; t < 2; ++t) {
    int jt = j0 + t * 16;
    v16bf kb0 = load_frag(kvb, 64, jt + r, 0, half);
    v16bf kb1 = load_frag(kvb, 64, jt + r, 32, half);
    v8f acc = {};
    acc = WMMA_BF16(qa0, kb0, acc);
    acc = WMMA_BF16(qa1, kb1, acc);
    if (MASKED) {
      int j = jt + r;                       // column held by this lane
#pragma unroll
      for (int v = 0; v < 8; ++v) {
        int qi = q0 + v + 8 * half;         // query row of element v
        if (!((j <= qi + 1) && (j <= Ntok))) acc[v] = -3.0e38f;
      }
    }
    s[t] = acc;
  }
  if (!MASKED) {  // keep next kv tile warm; reused by all heads/query blocks
    __builtin_prefetch(kvb + (size_t)(j0 + 32 + r) * 64, 0, 3);
  }

  // online softmax: row max via butterflies, P tile -> per-wave LDS region
  float crow[8];
#pragma unroll
  for (int v = 0; v < 8; ++v) {
    float tmax = fmaxf(s[0][v], s[1][v]);
#pragma unroll
    for (int m = 1; m < 16; m <<= 1) tmax = fmaxf(tmax, __shfl_xor(tmax, m, 32));
    float mnew = fmaxf(mrow[v], tmax);
    float c  = __expf(mrow[v] - mnew);
    float p0 = __expf(s[0][v] - mnew);
    float p1 = __expf(s[1][v] - mnew);
    crow[v] = c;
    mrow[v] = mnew;
#pragma unroll
    for (int t = 0; t < 4; ++t) o[t][v] *= c;
    plds[wave][v + 8 * half][r]      = (bf16)p0;
    plds[wave][v + 8 * half][16 + r] = (bf16)p1;
  }

  // per-wave LDS fence (same-wave DS ops are in-order; wait them to 0)
  asm volatile("s_wait_dscnt 0x0" ::: "memory");

  // read P back as 16x32 bf16 A-fragment
  v16bf pa;
  {
    const bf16* prow = &plds[wave][r][half * 8];
    v8bf lo = *(const v8bf*)(prow);
    v8bf hi = *(const v8bf*)(prow + 16);
#pragma unroll
    for (int i = 0; i < 8; ++i) { pa[i] = lo[i]; pa[8 + i] = hi[i]; }
  }

  // row sums of P via WMMA against ones: element v holds rowsum in all lanes
  {
    v8f rs = {};
    rs = WMMA_BF16(pa, ones, rs);
#pragma unroll
    for (int v = 0; v < 8; ++v) lrow[v] = lrow[v] * crow[v] + rs[v];
  }

  // O[16,64] += P[16,32] @ V[32,64]  (B operand from kvT, d-major)
#pragma unroll
  for (int t = 0; t < 4; ++t) {
    v16bf vb = load_frag(kvt, NPAD, t * 16 + r, j0, half);
    o[t] = WMMA_BF16(pa, vb, o[t]);
  }
}

// Grid: (Ntok/64, H, B); 4 waves/block, each wave owns 16 query rows.
__global__ __launch_bounds__(128) void flash_attn(const bf16* __restrict__ Q,   // [B*N,1024] pre-scaled
                                                  const bf16* __restrict__ KV,  // [B,NPAD,64]
                                                  const bf16* __restrict__ KVT, // [B,64,NPAD]
                                                  bf16* __restrict__ O,         // [B*N,1024]
                                                  int Ntok) {
  const int lane = threadIdx.x & 31;
  const int wave = threadIdx.x >> 5;
  const int r = lane & 15, half = lane >> 4;
  const int h  = blockIdx.y;
  const int b  = blockIdx.z;
  const int q0 = blockIdx.x * 64 + wave * 16;

  __shared__ bf16 plds[4][16][32];   // per-wave P staging tile

  const bf16* qbase = Q + ((size_t)(b * Ntok + q0)) * 1024 + h * 64;
  v16bf qa0 = load_frag(qbase, 1024, r, 0, half);
  v16bf qa1 = load_frag(qbase, 1024, r, 32, half);

  const bf16* kvb = KV  + (size_t)b * NPAD * 64;
  const bf16* kvt = KVT + (size_t)b * 64 * NPAD;

  v16bf ones;                        // B operand of the row-sum WMMA
#pragma unroll
  for (int i = 0; i < 16; ++i) ones[i] = (bf16)1.0f;

  v8f o[4] = {};
  float mrow[8], lrow[8];
#pragma unroll
  for (int v = 0; v < 8; ++v) { mrow[v] = -3.0e38f; lrow[v] = 0.0f; }

  // Unmasked steady state: a 32-col step is mask-free iff j0+31 <= q0+1
  // (min row in this wave's tile); the j<=Ntok bound then holds trivially.
  const int jlin = q0 - 30;
  const int jend = q0 + 16;          // max attended column for this wave
  int j0 = 0;
  for (; j0 <= jlin; j0 += 32)
    attn_step<false>(j0, q0, r, half, wave, Ntok, kvb, kvt, qa0, qa1, ones, o, mrow, lrow, plds);
  for (; j0 <= jend; j0 += 32)
    attn_step<true>(j0, q0, r, half, wave, Ntok, kvb, kvt, qa0, qa1, ones, o, mrow, lrow, plds);

  // normalize and store [b, q, h*64 + d] as bf16
  bf16* obase = O + ((size_t)(b * Ntok + q0)) * 1024 + h * 64;
#pragma unroll
  for (int v = 0; v < 8; ++v) {
    float inv = 1.0f / lrow[v];
#pragma unroll
    for (int t = 0; t < 4; ++t)
      obase[(size_t)(v + 8 * half) * 1024 + t * 16 + r] = (bf16)(o[t][v] * inv);
  }
}

// ------------------------------- launcher ----------------------------------

extern "C" void kernel_launch(void* const* d_in, const int* in_sizes, int n_in,
                              void* d_out, int out_size, void* d_ws, size_t ws_size,
                              hipStream_t stream) {
  (void)in_sizes; (void)n_in; (void)out_size; (void)ws_size;
  const float* x      = (const float*)d_in[0];
  const float* Wq     = (const float*)d_in[1];
  const float* Wkv    = (const float*)d_in[2];
  const float* nullkv = (const float*)d_in[3];
  const float* Wout   = (const float*)d_in[4];

  const int B = 4, Ntok = 2048, DIM = 1024, HD = 64, H = 16;
  const long BT = (long)B * Ntok;                 // 8192 token rows

  char* ws = (char*)d_ws;
  size_t off = 0;
  auto alloc = [&](size_t bytes) -> void* {
    void* p = ws + off;
    off += (bytes + 255) & ~(size_t)255;
    return p;
  };
  bf16* xb    = (bf16*)alloc((size_t)BT * DIM * 2);        // x in bf16
  bf16* WqT   = (bf16*)alloc((size_t)DIM * DIM * 2);       // Wq^T * (1/8)
  bf16* WkvT  = (bf16*)alloc((size_t)HD * DIM * 2);        // Wkv^T
  bf16* WoutT = (bf16*)alloc((size_t)DIM * DIM * 2);       // Wout^T
  bf16* qb    = (bf16*)alloc((size_t)BT * DIM * 2);        // q (pre-scaled)
  bf16* kvtmp = (bf16*)alloc((size_t)BT * HD * 2);         // x@Wkv
  bf16* kvb   = (bf16*)alloc((size_t)B * NPAD * HD * 2);   // kv, null + pad
  bf16* kvT   = (bf16*)alloc((size_t)B * HD * NPAD * 2);   // kv transposed
  bf16* ab    = (bf16*)alloc((size_t)BT * DIM * 2);        // attention out

  long nx = BT * DIM;
  cvt_f32_bf16<<<(unsigned)((nx + 255) / 256), 256, 0, stream>>>(x, xb, nx);
  transpose_cvt<<<(DIM * DIM + 255) / 256, 256, 0, stream>>>(Wq, WqT, DIM, DIM, 0.125f);
  transpose_cvt<<<(DIM * HD + 255) / 256, 256, 0, stream>>>(Wkv, WkvT, DIM, HD, 1.0f);
  transpose_cvt<<<(DIM * DIM + 255) / 256, 256, 0, stream>>>(Wout, WoutT, DIM, DIM, 1.0f);

  gemm_bf16<bf16><<<dim3(DIM / 64, (unsigned)(BT / 128)), 128, 0, stream>>>(xb, WqT, qb, (int)BT, DIM, DIM);
  gemm_bf16<bf16><<<dim3(HD / 64, (unsigned)(BT / 128)), 128, 0, stream>>>(xb, WkvT, kvtmp, (int)BT, HD, DIM);

  int nrep = B * NPAD * HD;
  repack_kv<<<(nrep + 255) / 256, 256, 0, stream>>>(kvtmp, nullkv, kvb, kvT, B, Ntok);

  flash_attn<<<dim3(Ntok / 64, H, B), 128, 0, stream>>>(qb, kvb, kvT, ab, Ntok);

  gemm_bf16<float><<<dim3(DIM / 64, (unsigned)(BT / 128)), 128, 0, stream>>>(ab, WoutT, (float*)d_out, (int)BT, DIM, DIM);
}
